// WindowAttention_11132555231834
// MI455X (gfx1250) — matrix operand order repaired and checked by reading, not compiled, not run on hardware
//
#include <hip/hip_runtime.h>

typedef __attribute__((ext_vector_type(16))) _Float16 v16h;
typedef __attribute__((ext_vector_type(8)))  float    v8f;

#define DIMC  384
#define HEADS 12
#define HD    32
#define NTOK  49
#define NPAD  64
#define BWIN  2048
#define NWIN  64
#define MROWS (BWIN * NTOK)   // 100352 = 6272 * 16
#define NSQ   (NTOK * NTOK)   // 2401

// ---------------------------------------------------------------------------
// Prep 1: elementwise f32 -> f16 conversion (x, qkv_w, proj_w).
// ---------------------------------------------------------------------------
__global__ __launch_bounds__(256) void cvt_f16_kernel(
    const float* __restrict__ src, _Float16* __restrict__ dst, int n)
{
    const int i = blockIdx.x * 256 + threadIdx.x;
    if (i < n) dst[i] = (_Float16)src[i];
}

// ---------------------------------------------------------------------------
// Prep 2: fused bias+mask table  bm[w][h][i][j] =
//         rel_table[rel_index[i,j], h] + mask[w, i, j]   (64*12*49*49 f32).
// ---------------------------------------------------------------------------
__global__ __launch_bounds__(256) void build_bm_kernel(
    const float* __restrict__ mask, const float* __restrict__ rel_table,
    const int* __restrict__ rel_index, float* __restrict__ bm)
{
    const int tid = blockIdx.x * 256 + threadIdx.x;
    const int total = NWIN * HEADS * NSQ;
    if (tid >= total) return;
    const int ij = tid % NSQ;
    const int h  = (tid / NSQ) % HEADS;
    const int w  = tid / (NSQ * HEADS);
    bm[tid] = rel_table[rel_index[ij] * HEADS + h] + mask[w * NSQ + ij];
}

// ---------------------------------------------------------------------------
// Kernel 1: QKV projection.  C[100352,1152] = x16 @ qkv_w16^T + b.
// One wave computes a 16-row x 128-col strip (8 WMMA tiles, A-frag reused).
// All operands f16, loads are contiguous 16B/32B -> global_load_b128.
// q scaled by hd^-0.5 -> qh[b,h,n,d]; k -> kh[b,h,n,d]; v -> vT[b,h,d,n].
// ---------------------------------------------------------------------------
__global__ __launch_bounds__(256) void swin_qkv_kernel(
    const _Float16* __restrict__ x, const _Float16* __restrict__ w,
    const float* __restrict__ bqkv,
    _Float16* __restrict__ qh, _Float16* __restrict__ kh,
    _Float16* __restrict__ vT)
{
    const int wave = threadIdx.x >> 5;
    const int lane = threadIdx.x & 31;
    const int wid  = blockIdx.x * 8 + wave;
    const int rowTile = wid / 9;          // 1152/128 = 9 col groups
    const int cg      = wid % 9;
    if (rowTile >= MROWS / 16) return;    // uniform; grid is exact
    const int m0   = rowTile * 16;
    const int n0   = cg * 128;
    const int half = lane >> 4;
    const int l16  = lane & 15;

    v8f acc[8] = {};
    const _Float16* arow = x + (size_t)(m0 + l16) * DIMC;

    for (int k0 = 0; k0 < DIMC; k0 += 32) {
        v16h afrag;
        #pragma unroll
        for (int j = 0; j < 8; ++j) {
            afrag[j]     = arow[k0 + half * 8 + j];
            afrag[j + 8] = arow[k0 + 16 + half * 8 + j];
        }
        #pragma unroll
        for (int t = 0; t < 8; ++t) {
            const int col = n0 + t * 16 + l16;
            const _Float16* wrow = w + (size_t)col * DIMC + k0 + half * 8;
            v16h bfrag;
            #pragma unroll
            for (int j = 0; j < 16; ++j) bfrag[j] = wrow[j + (j >= 8 ? 8 : 0)];
            acc[t] = __builtin_amdgcn_wmma_f32_16x16x32_f16(
                false, afrag, false, bfrag, (short)0, acc[t], false, false);
        }
    }

    const float scale = 0.17677669529663687f;  // 32^-0.5
    #pragma unroll
    for (int t = 0; t < 8; ++t) {
        const int col   = n0 + t * 16 + l16;
        const int which = col / DIMC;
        const int h     = (col % DIMC) / HD;
        const int d     = col % HD;
        const float bb  = bqkv[col];
        #pragma unroll
        for (int r = 0; r < 8; ++r) {
            const int gr = m0 + r + half * 8;
            const int b  = gr / NTOK, n = gr % NTOK;
            const int bh = b * HEADS + h;
            const float v = acc[t][r] + bb;
            if (which == 0)
                qh[((size_t)bh * NTOK + n) * HD + d] = (_Float16)(v * scale);
            else if (which == 1)
                kh[((size_t)bh * NTOK + n) * HD + d] = (_Float16)v;
            else
                vT[((size_t)bh * HD + d) * NTOK + n] = (_Float16)v;
        }
    }
}

// ---------------------------------------------------------------------------
// Kernel 2: per-(window, head) attention.  One wave per (b, h).
// S = q @ k^T (49 padded to 64, 4x4 WMMA tiles, K=32), + fused bias/mask
// table (single f32 load per element), row softmax via shfl_xor inside
// 16-lane groups, P staged as f16 in per-wave LDS (C-layout -> A-layout
// transpose), then out = P @ v (K=64, 2 k-steps of 32).
// ---------------------------------------------------------------------------
__global__ __launch_bounds__(128) void swin_attn_kernel(
    const _Float16* __restrict__ qh, const _Float16* __restrict__ kh,
    const _Float16* __restrict__ vT, const float* __restrict__ bm,
    _Float16* __restrict__ ao)
{
    __shared__ _Float16 lds[4][NPAD * NPAD];   // 8 KB per wave, 32 KB total
    const int wave = threadIdx.x >> 5;
    const int lane = threadIdx.x & 31;
    const int wid  = blockIdx.x * 4 + wave;
    const int b    = wid / HEADS;
    const int h    = wid % HEADS;
    const int half = lane >> 4;
    const int l16  = lane & 15;
    const size_t qkbase = (size_t)(b * HEADS + h) * NTOK * HD;

    // Q A-fragments (rows padded with zero)
    v16h aq[4];
    #pragma unroll
    for (int i = 0; i < 4; ++i) {
        const int row = i * 16 + l16;
        const _Float16* qrow = qh + qkbase + (size_t)row * HD;
        #pragma unroll
        for (int j = 0; j < 8; ++j) {
            aq[i][j]     = (row < NTOK) ? qrow[half * 8 + j]      : (_Float16)0.f;
            aq[i][j + 8] = (row < NTOK) ? qrow[16 + half * 8 + j] : (_Float16)0.f;
        }
    }
    // K^T B-fragments (d = half*8 + t, then 16 + half*8 + t-8, matching A)
    v16h bk[4];
    #pragma unroll
    for (int j = 0; j < 4; ++j) {
        const int col = j * 16 + l16;
        const _Float16* krow = kh + qkbase + (size_t)col * HD + half * 8;
        #pragma unroll
        for (int t = 0; t < 16; ++t)
            bk[j][t] = (col < NTOK) ? krow[t + (t >= 8 ? 8 : 0)] : (_Float16)0.f;
    }

    v8f s[4][4] = {};
    #pragma unroll
    for (int i = 0; i < 4; ++i)
        #pragma unroll
        for (int j = 0; j < 4; ++j)
            s[i][j] = __builtin_amdgcn_wmma_f32_16x16x32_f16(
                false, aq[i], false, bk[j], (short)0, s[i][j], false, false);

    // fused bias+mask + softmax, write P (f16) to LDS
    const float* bmrow = bm + ((size_t)(b & (NWIN - 1)) * HEADS + h) * NSQ;
    #pragma unroll
    for (int i = 0; i < 4; ++i) {
        #pragma unroll
        for (int r = 0; r < 8; ++r) {
            const int gi = i * 16 + r + half * 8;
            float e[4];
            #pragma unroll
            for (int j = 0; j < 4; ++j) {
                const int gj = j * 16 + l16;
                float v = s[i][j][r];
                if (gi < NTOK && gj < NTOK)
                    v += bmrow[gi * NTOK + gj];
                else
                    v = -1e30f;   // pad cols -> 0 after softmax
                e[j] = v;
            }
            float m = fmaxf(fmaxf(e[0], e[1]), fmaxf(e[2], e[3]));
            for (int d = 1; d < 16; d <<= 1) m = fmaxf(m, __shfl_xor(m, d, 32));
            float sum = 0.f;
            #pragma unroll
            for (int j = 0; j < 4; ++j) { e[j] = __expf(e[j] - m); sum += e[j]; }
            for (int d = 1; d < 16; d <<= 1) sum += __shfl_xor(sum, d, 32);
            const float inv = 1.f / sum;
            #pragma unroll
            for (int j = 0; j < 4; ++j)
                lds[wave][gi * NPAD + j * 16 + l16] = (_Float16)(e[j] * inv);
        }
    }
    __syncthreads();

    // out = P @ v  (K = 64 -> 2 k-steps of 32; 2 col tiles of 16)
    const size_t vbase = (size_t)(b * HEADS + h) * HD * NTOK;
    v8f o[4][2] = {};
    #pragma unroll
    for (int kt = 0; kt < 2; ++kt) {
        v16h bv[2];
        #pragma unroll
        for (int dt = 0; dt < 2; ++dt) {
            const int d = dt * 16 + l16;
            const _Float16* vrow = vT + vbase + (size_t)d * NTOK;
            #pragma unroll
            for (int t = 0; t < 16; ++t) {
                const int gj = kt * 32 + half * 8 + t + (t >= 8 ? 8 : 0);
                bv[dt][t] = (gj < NTOK) ? vrow[gj] : (_Float16)0.f;
            }
        }
        #pragma unroll
        for (int i = 0; i < 4; ++i) {
            const int row = i * 16 + l16;
            v16h ap;
            #pragma unroll
            for (int j = 0; j < 8; ++j) {
                ap[j]     = lds[wave][row * NPAD + kt * 32 + half * 8 + j];
                ap[j + 8] = lds[wave][row * NPAD + kt * 32 + 16 + half * 8 + j];
            }
            #pragma unroll
            for (int dt = 0; dt < 2; ++dt)
                o[i][dt] = __builtin_amdgcn_wmma_f32_16x16x32_f16(
                    false, ap, false, bv[dt], (short)0, o[i][dt], false, false);
        }
    }

    #pragma unroll
    for (int i = 0; i < 4; ++i)
        #pragma unroll
        for (int dt = 0; dt < 2; ++dt)
            #pragma unroll
            for (int r = 0; r < 8; ++r) {
                const int gi = i * 16 + r + half * 8;
                if (gi < NTOK)
                    ao[((size_t)(b * NTOK + gi)) * DIMC + h * HD + dt * 16 + l16]
                        = (_Float16)o[i][dt][r];
            }
}

// ---------------------------------------------------------------------------
// Kernel 3: output projection.  out[100352,384] = ao @ proj_w16^T + proj_b.
// Same skeleton as kernel 1; all operands f16; fp32 output + bias.
// ---------------------------------------------------------------------------
__global__ __launch_bounds__(256) void swin_proj_kernel(
    const _Float16* __restrict__ a, const _Float16* __restrict__ w,
    const float* __restrict__ bias, float* __restrict__ out)
{
    const int wave = threadIdx.x >> 5;
    const int lane = threadIdx.x & 31;
    const int wid  = blockIdx.x * 8 + wave;
    const int rowTile = wid / 3;          // 384/128 = 3 col groups
    const int cg      = wid % 3;
    if (rowTile >= MROWS / 16) return;
    const int m0   = rowTile * 16;
    const int n0   = cg * 128;
    const int half = lane >> 4;
    const int l16  = lane & 15;

    v8f acc[8] = {};
    const _Float16* arow = a + (size_t)(m0 + l16) * DIMC;

    for (int k0 = 0; k0 < DIMC; k0 += 32) {
        v16h afrag;
        #pragma unroll
        for (int j = 0; j < 8; ++j) {
            afrag[j]     = arow[k0 + half * 8 + j];
            afrag[j + 8] = arow[k0 + 16 + half * 8 + j];
        }
        #pragma unroll
        for (int t = 0; t < 8; ++t) {
            const int col = n0 + t * 16 + l16;
            const _Float16* wrow = w + (size_t)col * DIMC + k0 + half * 8;
            v16h bfrag;
            #pragma unroll
            for (int j = 0; j < 16; ++j) bfrag[j] = wrow[j + (j >= 8 ? 8 : 0)];
            acc[t] = __builtin_amdgcn_wmma_f32_16x16x32_f16(
                false, afrag, false, bfrag, (short)0, acc[t], false, false);
        }
    }

    #pragma unroll
    for (int t = 0; t < 8; ++t) {
        const int col = n0 + t * 16 + l16;
        const float bb = bias[col];
        #pragma unroll
        for (int r = 0; r < 8; ++r) {
            const int gr = m0 + r + half * 8;
            out[(size_t)gr * DIMC + col] = acc[t][r] + bb;
        }
    }
}

// ---------------------------------------------------------------------------
extern "C" void kernel_launch(void* const* d_in, const int* in_sizes, int n_in,
                              void* d_out, int out_size, void* d_ws, size_t ws_size,
                              hipStream_t stream) {
    const float* x         = (const float*)d_in[0];
    const float* mask      = (const float*)d_in[1];
    const float* qkv_w     = (const float*)d_in[2];
    const float* qkv_b     = (const float*)d_in[3];
    const float* proj_w    = (const float*)d_in[4];
    const float* proj_b    = (const float*)d_in[5];
    const float* rel_table = (const float*)d_in[6];
    const int*   rel_index = (const int*)d_in[7];
    float* out = (float*)d_out;

    // ---- workspace layout (halves unless noted) --------------------------
    const size_t Q      = (size_t)BWIN * HEADS * NTOK * HD;  // 38,535,168
    const size_t NX     = (size_t)MROWS * DIMC;              // 38,535,168
    const size_t NQKVW  = (size_t)3 * DIMC * DIMC;           //    442,368
    const size_t NPROJW = (size_t)DIMC * DIMC;               //    147,456
    const size_t NBM    = (size_t)NWIN * HEADS * NSQ;        //  1,843,968 (f32)

    _Float16* ws   = (_Float16*)d_ws;
    _Float16* qh   = ws;
    _Float16* kh   = ws + Q;
    _Float16* vT   = ws + 2 * Q;
    _Float16* ao   = ws + 3 * Q;
    _Float16* x16  = ws + 4 * Q;
    _Float16* qw16 = x16 + NX;
    _Float16* pw16 = qw16 + NQKVW;
    float*    bm   = (float*)(pw16 + NPROJW);   // even half-offset -> aligned

    // ---- prep: f16 operands + fused bias/mask table ----------------------
    cvt_f16_kernel<<<(int)((NX + 255) / 256), 256, 0, stream>>>(x, x16, (int)NX);
    cvt_f16_kernel<<<(int)((NQKVW + 255) / 256), 256, 0, stream>>>(qkv_w, qw16, (int)NQKVW);
    cvt_f16_kernel<<<(int)((NPROJW + 255) / 256), 256, 0, stream>>>(proj_w, pw16, (int)NPROJW);
    build_bm_kernel<<<(int)((NBM + 255) / 256), 256, 0, stream>>>(mask, rel_table, rel_index, bm);

    // ---- main pipeline ---------------------------------------------------
    // 6272 row tiles * 9 col groups = 56448 waves / 8 per block
    swin_qkv_kernel<<<7056, 256, 0, stream>>>(x16, qw16, qkv_b, qh, kh, vT);
    // 2048 * 12 = 24576 (b,h) waves / 4 per block
    swin_attn_kernel<<<6144, 128, 0, stream>>>(qh, kh, vT, bm, ao);
    // 6272 row tiles * 3 col groups = 18816 waves / 8 per block
    swin_proj_kernel<<<2352, 256, 0, stream>>>(ao, pw16, proj_b, out);
}
